// LocalAttention_26972394619677
// MI455X (gfx1250) — matrix operand order repaired and checked by reading, not compiled, not run on hardware
//
#include <hip/hip_runtime.h>
#include <stdint.h>

#define B_ 4
#define S_ 4096
#define H_ 64

typedef __attribute__((ext_vector_type(16))) __bf16 v16bf;
typedef __attribute__((ext_vector_type(8)))  float  v8f;
typedef __attribute__((ext_vector_type(4)))  unsigned int u32x4;
typedef __attribute__((ext_vector_type(8)))  int    i32x8;
typedef __attribute__((ext_vector_type(4)))  int    i32x4;

union FragBF {
  v16bf v;
  unsigned short u[16];
  u32x4 q[2];
};

__device__ __forceinline__ unsigned short f2bf(float x) {
  unsigned u = __builtin_bit_cast(unsigned, x);
  u += 0x7FFFu + ((u >> 16) & 1u);     // round-to-nearest-even
  return (unsigned short)(u >> 16);
}

// max-reduce across each 16-lane row via DPP16 butterfly (xor1,xor2,xor7,xor15)
__device__ __forceinline__ float dpp_max16(float x) {
  int t;
  t = __builtin_amdgcn_update_dpp(0, __builtin_bit_cast(int, x), 0xB1, 0xf, 0xf, true);  // quad_perm [1,0,3,2]
  x = fmaxf(x, __builtin_bit_cast(float, t));
  t = __builtin_amdgcn_update_dpp(0, __builtin_bit_cast(int, x), 0x4E, 0xf, 0xf, true);  // quad_perm [2,3,0,1]
  x = fmaxf(x, __builtin_bit_cast(float, t));
  t = __builtin_amdgcn_update_dpp(0, __builtin_bit_cast(int, x), 0x141, 0xf, 0xf, true); // row_half_mirror
  x = fmaxf(x, __builtin_bit_cast(float, t));
  t = __builtin_amdgcn_update_dpp(0, __builtin_bit_cast(int, x), 0x140, 0xf, 0xf, true); // row_mirror
  x = fmaxf(x, __builtin_bit_cast(float, t));
  return x;
}

// ---------------- Kernel 1: Q/K/V projections (fp32 math) ----------------
// q is pre-scaled by log2(e)/sqrt(64) so the softmax can use exp2.
__global__ __launch_bounds__(256) void proj_kernel(
    const float* __restrict__ feat,
    const float* __restrict__ Wq, const float* __restrict__ bq,
    const float* __restrict__ Wk, const float* __restrict__ bk,
    const float* __restrict__ Wv, const float* __restrict__ bv,
    unsigned short* __restrict__ qbf,
    float* __restrict__ kf,
    unsigned short* __restrict__ vbf)
{
  const float QSCALE = 0.125f * 1.4426950408889634f;
  int rb = blockIdx.x * 4;                 // 4 sequence rows per block
  int r  = threadIdx.x >> 6;
  int h  = threadIdx.x & 63;
  __shared__ float fs[4][64];
  fs[r][h] = feat[(size_t)(rb + r) * 64 + h];
  __syncthreads();
  float q = bq[h], k = bk[h], v = bv[h];
#pragma unroll 8
  for (int d = 0; d < 64; ++d) {
    float f = fs[r][d];
    q = fmaf(f, Wq[h * 64 + d], q);
    k = fmaf(f, Wk[h * 64 + d], k);
    v = fmaf(f, Wv[h * 64 + d], v);
  }
  size_t row = (size_t)(rb + r) * 64 + h;
  qbf[row] = f2bf(q * QSCALE);
  kf[row]  = k;
  vbf[row] = f2bf(v);
}

// ---------------- Kernel 2: per-(b,h) column sum of K ----------------
__global__ __launch_bounds__(256) void ksum_kernel(
    const float* __restrict__ kf, float* __restrict__ ksum)
{
  int b = blockIdx.x;
  int h = threadIdx.x & 63;
  int stripe = threadIdx.x >> 6;           // 4 stripes of S/4 rows
  const float* p = kf + ((size_t)b * S_ + (size_t)stripe * (S_ / 4)) * 64 + h;
  float acc = 0.f;
  for (int s = 0; s < S_ / 4; ++s) acc += p[(size_t)s * 64];
  __shared__ float red[256];
  red[threadIdx.x] = acc;
  __syncthreads();
  if (stripe == 0)
    ksum[b * 64 + h] = red[h] + red[64 + h] + red[128 + h] + red[192 + h];
}

// ---------------- Kernel 3: mixed_k = (sum_all + (e-1)*band_sum)/Z_s ----------------
__global__ __launch_bounds__(256) void mixedk_kernel(
    const float* __restrict__ kf, const float* __restrict__ ksum,
    unsigned short* __restrict__ mkbf)
{
  int grow = blockIdx.x * 4 + (threadIdx.x >> 6);   // global row in [B*S]
  int h = threadIdx.x & 63;
  int b = grow >> 12;
  int s = grow & (S_ - 1);
  int lo = s - 2; if (lo < 0) lo = 0;
  int hi = s + 2; if (hi > S_ - 1) hi = S_ - 1;
  float bs = 0.f;
  for (int j = lo; j <= hi; ++j) bs += kf[((size_t)b * S_ + j) * 64 + h];
  const float EM1 = 1.7182818284590452f;            // e - 1
  float mk = (ksum[b * 64 + h] + EM1 * bs) /
             ((float)S_ + EM1 * (float)(hi - lo + 1));
  mkbf[(size_t)grow * 64 + h] = f2bf(mk);
}

// ---------------- Kernel 4: flash attention with bf16 WMMA + TDM ----------------
// Block = 128 threads (4 waves). Each wave owns 16 query rows; block covers 64.
// K tile (32x64 bf16) staged in LDS via TENSOR_LOAD_TO_LDS; V tile transposed by VALU.
__global__ __launch_bounds__(128) void flash_kernel(
    const unsigned short* __restrict__ qbf,
    const unsigned short* __restrict__ mkbf,
    const unsigned short* __restrict__ vbf,
    float* __restrict__ out)
{
  int b    = blockIdx.x >> 6;              // 64 query blocks per batch
  int qblk = blockIdx.x & 63;
  int wave = threadIdx.x >> 5;
  int lane = threadIdx.x & 31;
  int hl   = lane >> 4;                    // half-wave (0/1)
  int ln   = lane & 15;
  int qbase = qblk * 64 + wave * 16;

  __shared__ __align__(16) unsigned short kt[32 * 64];     // K tile [kv][d]
  __shared__ __align__(16) unsigned short vt[64 * 32];     // V tile transposed [d][kv]
  __shared__ __align__(16) unsigned short pst[4][16 * 32]; // per-wave P staging [m][kv]

  unsigned ktLds = (unsigned)(uintptr_t)(&kt[0]);          // LDS byte offset (addr[31:0])

  // --- Q fragments (A-matrix 16x32 bf16 layout), loaded once ---
  FragBF qa[2];
  {
    const unsigned short* qrow =
        qbf + ((size_t)(b * S_ + qbase + ln)) * 64 + 16 * hl;
#pragma unroll
    for (int dh = 0; dh < 2; ++dh) {
      qa[dh].q[0] = *(const u32x4*)(qrow + dh * 32);
      qa[dh].q[1] = *(const u32x4*)(qrow + dh * 32 + 8);
    }
  }

  // all-ones bf16 B fragment for row-sum WMMA
  FragBF ones;
#pragma unroll
  for (int i = 0; i < 2; ++i)
#pragma unroll
    for (int j = 0; j < 4; ++j) ones.q[i][j] = 0x3F803F80u;

  v8f oacc[4];
#pragma unroll
  for (int t = 0; t < 4; ++t)
#pragma unroll
    for (int i = 0; i < 8; ++i) oacc[t][i] = 0.f;
  v8f lacc;                                  // running softmax denominators (all N equal)
#pragma unroll
  for (int i = 0; i < 8; ++i) lacc[i] = 0.f;

  float mrow[8];
#pragma unroll
  for (int r = 0; r < 8; ++r) mrow[r] = -1e30f;

  for (int kv0 = 0; kv0 < S_; kv0 += 32) {
    __syncthreads();                        // previous tile fully consumed

    // --- K tile: Tensor Data Mover, issued by wave 0 ---
    if (wave == 0) {
      unsigned long long ga =
          (unsigned long long)(uintptr_t)(mkbf + ((size_t)(b * S_ + kv0)) * 64);
      u32x4 g0;
      g0[0] = 1u;                                   // count=1, user mode
      g0[1] = ktLds;                                // lds_addr
      g0[2] = (unsigned)ga;                         // global_addr[31:0]
      g0[3] = (unsigned)(ga >> 32) | (2u << 30);    // global_addr[56:32] | type=2
      i32x8 g1;
      g1[0] = 0x00010000;                           // data_size=1 (2 bytes)
      g1[1] = 64 << 16;                             // tensor_dim0 = 64
      g1[2] = (int)(4096u << 16);                   // tensor_dim1 = 4096
      g1[3] = 64 << 16;                             // tile_dim0 = 64
      g1[4] = 32;                                   // tile_dim1 = 32
      g1[5] = 64;                                   // tensor_dim0_stride = 64
      g1[6] = 0;
      g1[7] = 0;
      i32x4 gz4 = {0, 0, 0, 0};
      i32x8 gz8 = {0, 0, 0, 0, 0, 0, 0, 0};
      __builtin_amdgcn_tensor_load_to_lds(g0, g1, gz4, gz4, gz8, 0);
    }

    // --- V tile: cooperative transposed staging (packed b32 stores) ---
    {
      const unsigned short* vsrc = vbf + ((size_t)(b * S_ + kv0)) * 64;
      int c   = threadIdx.x;     // 128 threads = 16 kv-pairs x 8 d-chunks
      int kv2 = c >> 3;          // kv pair: rows 2*kv2, 2*kv2+1
      int d8  = (c & 7) * 8;
      u32x4 r0 = *(const u32x4*)(vsrc + (size_t)(2 * kv2) * 64 + d8);
      u32x4 r1 = *(const u32x4*)(vsrc + (size_t)(2 * kv2 + 1) * 64 + d8);
      unsigned* vt32 = (unsigned*)vt;
#pragma unroll
      for (int i = 0; i < 4; ++i) {
        unsigned a = r0[i], bb = r1[i];
        vt32[(d8 + 2 * i) * 16 + kv2]     = (a & 0xffffu) | (bb << 16);
        vt32[(d8 + 2 * i + 1) * 16 + kv2] = (a >> 16) | (bb & 0xffff0000u);
      }
    }
    if (wave == 0) __builtin_amdgcn_s_wait_tensorcnt(0);
    __syncthreads();                        // kt/vt ready for all waves

    // --- scores: S[16q x 32kv] = Q(16x64) @ K^T(64x32), two accs by kv-half
    v8f sacc[2];
#pragma unroll
    for (int j = 0; j < 2; ++j)
#pragma unroll
      for (int i = 0; i < 8; ++i) sacc[j][i] = 0.f;

#pragma unroll
    for (int dh = 0; dh < 2; ++dh) {
#pragma unroll
      for (int j = 0; j < 2; ++j) {
        FragBF kb;  // B-frag (32x16): N = j*16+ln; elem e -> K = e + 16*hl
        const unsigned short* src = &kt[(j * 16 + ln) * 64 + dh * 32 + 16 * hl];
        kb.q[0] = *(const u32x4*)(src);
        kb.q[1] = *(const u32x4*)(src + 8);
        sacc[j] = __builtin_amdgcn_wmma_f32_16x16x32_bf16(
            false, qa[dh].v, false, kb.v, (short)0, sacc[j], false, false);
      }
    }

    // --- online softmax; C layout: vgpr r, lane -> row M = r + 8*hl, N = ln
#pragma unroll
    for (int r = 0; r < 8; ++r) {
      float s0 = sacc[0][r], s1 = sacc[1][r];
      float mx    = dpp_max16(fmaxf(s0, s1));
      float mnew  = fmaxf(mrow[r], mx);
      float alpha = __builtin_amdgcn_exp2f(mrow[r] - mnew);
      mrow[r] = mnew;
      float p0 = __builtin_amdgcn_exp2f(s0 - mnew);
      float p1 = __builtin_amdgcn_exp2f(s1 - mnew);
      int mr = r + 8 * hl;
      pst[wave][mr * 32 + ln]      = f2bf(p0);
      pst[wave][mr * 32 + 16 + ln] = f2bf(p1);
      lacc[r] *= alpha;
#pragma unroll
      for (int t = 0; t < 4; ++t) oacc[t][r] *= alpha;
    }
    // wave-private LDS staging: wait for this wave's DS ops (cross-lane reshuffle)
    asm volatile("s_wait_dscnt 0x0" ::: "memory");

    // P as A-fragment (16x32): lane row = ln; contiguous kv = 16*hl .. 16*hl+15
    FragBF pa;
    {
      const unsigned short* ps = &pst[wave][ln * 32 + 16 * hl];
      pa.q[0] = *(const u32x4*)(ps);
      pa.q[1] = *(const u32x4*)(ps + 8);
    }

    // row-sums: lacc += P @ ones  (every column = row sum)
    lacc = __builtin_amdgcn_wmma_f32_16x16x32_bf16(
        false, pa.v, false, ones.v, (short)0, lacc, false, false);

    // --- O += P(16x32) @ V(32x64), 4 d-tiles of 16
#pragma unroll
    for (int t = 0; t < 4; ++t) {
      FragBF vb;  // B-frag: N = t*16+ln (d), elem e -> K(kv) = e + 16*hl
      const unsigned short* src = &vt[(t * 16 + ln) * 32 + 16 * hl];
      vb.q[0] = *(const u32x4*)(src);
      vb.q[1] = *(const u32x4*)(src + 8);
      oacc[t] = __builtin_amdgcn_wmma_f32_16x16x32_bf16(
          false, pa.v, false, vb.v, (short)0, oacc[t], false, false);
    }
  }

  // --- normalize and store fp32 output
#pragma unroll
  for (int r = 0; r < 8; ++r) {
    float inv = 1.0f / lacc[r];
    int row = qbase + r + 8 * hl;
    float* orow = out + ((size_t)(b * S_ + row)) * 64 + ln;
#pragma unroll
    for (int t = 0; t < 4; ++t) orow[t * 16] = oacc[t][r] * inv;
  }
}

// ---------------- Host launcher ----------------
extern "C" void kernel_launch(void* const* d_in, const int* in_sizes, int n_in,
                              void* d_out, int out_size, void* d_ws, size_t ws_size,
                              hipStream_t stream) {
  const float* feat = (const float*)d_in[0];
  const float* Wq = (const float*)d_in[1];
  const float* bq = (const float*)d_in[2];
  const float* Wk = (const float*)d_in[3];
  const float* bk = (const float*)d_in[4];
  const float* Wv = (const float*)d_in[5];
  const float* bv = (const float*)d_in[6];
  float* out = (float*)d_out;

  char* ws = (char*)d_ws;
  unsigned short* qbf  = (unsigned short*)(ws);                        // 2 MB
  unsigned short* vbf  = (unsigned short*)(ws + (2ull << 20));         // 2 MB
  unsigned short* mkbf = (unsigned short*)(ws + (4ull << 20));         // 2 MB
  float*          kf   = (float*)(ws + (6ull << 20));                  // 4 MB
  float*          ksum = (float*)(ws + (10ull << 20));                 // 1 KB

  proj_kernel<<<(B_ * S_) / 4, 256, 0, stream>>>(feat, Wq, bq, Wk, bk, Wv, bv,
                                                 qbf, kf, vbf);
  ksum_kernel<<<B_, 256, 0, stream>>>(kf, ksum);
  mixedk_kernel<<<(B_ * S_) / 4, 256, 0, stream>>>(kf, ksum, mkbf);
  flash_kernel<<<B_ * (S_ / 64), 128, 0, stream>>>(qbf, mkbf, vbf, out);
}
